// AttentionHead_37752762532262
// MI455X (gfx1250) — matrix operand order repaired and checked by reading, not compiled, not run on hardware
//
#include <hip/hip_runtime.h>

// ---------------------------------------------------------------------------
// Causal flash-attention for MI455X gfx1250 (wave32, WMMA).
// B=16, S=2048, D=64, fp32 in/out, scale=1/64 (power of two -> folded exactly
// into the f16 Q staging), reference's "exact zero gets -1e20" quirk kept.
// Compute-bound: inner loop is 18x v_wmma_f32_16x16x32_f16 per 64-key tile
// (8 QK^T + 8 PV + 2 row-sum-via-ones), f32 accumulate + f32 online softmax.
// Causal masking is hoisted: interior tiles take a mask-free path, fully
// masked tiles skip compute entirely (wave-uniform branches, barriers stay
// workgroup-uniform). K/V staging is register-pipelined across tiles.
// ---------------------------------------------------------------------------

typedef __attribute__((ext_vector_type(16))) _Float16 v16h;
typedef __attribute__((ext_vector_type(8)))  _Float16 v8h;
typedef __attribute__((ext_vector_type(8)))  float    v8f;

static __device__ __forceinline__ v16h combine16(v8h lo, v8h hi) {
    v16h r;
#pragma unroll
    for (int i = 0; i < 8; ++i) { r[i] = lo[i]; r[i + 8] = hi[i]; }
    return r;
}

#define WMMA_F16(Aop, Bop, Cop) \
    __builtin_amdgcn_wmma_f32_16x16x32_f16(false, (Aop), false, (Bop), (short)0, (Cop), false, false)

__global__ __launch_bounds__(256) void
fa_causal_kernel(const float* __restrict__ q, const float* __restrict__ k,
                 const float* __restrict__ v, float* __restrict__ out) {
    constexpr int   S     = 2048, D = 64, QTILE = 128, KTILE = 64;
    constexpr float NEGV  = -1.0e20f;
    constexpr float SCALE = 1.0f / 64.0f;

    const int tileIdx = blockIdx.x % (S / QTILE);
    const int b       = blockIdx.x / (S / QTILE);
    const int q0      = tileIdx * QTILE;

    const int tid  = threadIdx.x;
    const int wave = tid >> 5;
    const int lane = tid & 31;
    const int half = lane >> 4;
    const int ln   = lane & 15;

    // Rows padded +8 halves (16B): conflict-avoiding, keeps v8h casts aligned.
    __shared__ __align__(16) _Float16 Qlds[QTILE][72];        // 18432 B
    __shared__ __align__(16) _Float16 Klds[KTILE][72];        //  9216 B
    __shared__ __align__(16) _Float16 Vtlds[D][KTILE + 8];    //  9216 B (V^T)
    __shared__ __align__(16) _Float16 Plds[8][16][KTILE + 8]; // 18432 B

    // ---- stage Q (scale folded in; exact since 1/64 is a power of two) ---
    {
        const int row = tid >> 1;            // 0..127
        const int col = (tid & 1) * 32;      // 0 or 32
        const float* src = q + ((size_t)b * S + q0 + row) * D + col;
#pragma unroll
        for (int c = 0; c < 4; ++c) {
            v8h h;
#pragma unroll
            for (int x = 0; x < 8; ++x) h[x] = (_Float16)(src[c * 8 + x] * SCALE);
            *(v8h*)&Qlds[row][col + c * 8] = h;
        }
    }
    __syncthreads();

    // ---- per-wave Q A-operands (16x32 f16 layout), resident in VGPRs -----
    const int qrow = wave * 16 + ln;
    v16h a0 = combine16(*(const v8h*)&Qlds[qrow][half * 8],
                        *(const v8h*)&Qlds[qrow][16 + half * 8]);      // d=0..31
    v16h a1 = combine16(*(const v8h*)&Qlds[qrow][32 + half * 8],
                        *(const v8h*)&Qlds[qrow][48 + half * 8]);      // d=32..63

    v16h onesB;                               // B = ones(32x16): row-sum WMMA
#pragma unroll
    for (int i = 0; i < 16; ++i) onesB[i] = (_Float16)1.0f;

    v8f o0 = {}, o1 = {}, o2 = {}, o3 = {};   // O accumulator (16x64 f32)
    v8f lC = {};                              // softmax denominator (per row)
    float m[8];
#pragma unroll
    for (int r = 0; r < 8; ++r) m[r] = -3.0e38f;

    const int iMin  = q0 + wave * 16;         // this wave's min/max query row
    const int iMax  = iMin + 15;
    const int ibase = iMin + half * 8;
    const int nkv   = (q0 + QTILE) / KTILE;   // causal loop bound

    // ---- register-pipelined K/V staging ----------------------------------
    const int srowi = tid >> 2;               // 0..63 (key row)
    const int scol  = (tid & 3) * 16;         // 0/16/32/48 (head dim)
    float kr[16], vr[16];
    auto load_tile = [&](int t) {
        const float* ks = k + ((size_t)b * S + t * KTILE + srowi) * D + scol;
        const float* vs = v + ((size_t)b * S + t * KTILE + srowi) * D + scol;
#pragma unroll
        for (int c = 0; c < 4; ++c) {
            float4 kt = ((const float4*)ks)[c];
            float4 vt = ((const float4*)vs)[c];
            kr[c*4+0]=kt.x; kr[c*4+1]=kt.y; kr[c*4+2]=kt.z; kr[c*4+3]=kt.w;
            vr[c*4+0]=vt.x; vr[c*4+1]=vt.y; vr[c*4+2]=vt.z; vr[c*4+3]=vt.w;
        }
    };
    load_tile(0);

    for (int t = 0; t < nkv; ++t) {
        const int kv0 = t * KTILE;

        __syncthreads();                      // prior tile's LDS reads done
#pragma unroll
        for (int c = 0; c < 2; ++c) {         // K row-major
            v8h h;
#pragma unroll
            for (int x = 0; x < 8; ++x) h[x] = (_Float16)kr[c * 8 + x];
            *(v8h*)&Klds[srowi][scol + c * 8] = h;
        }
#pragma unroll
        for (int x = 0; x < 16; ++x)          // V transposed
            Vtlds[scol + x][srowi] = (_Float16)vr[x];
        __syncthreads();

        if (t + 1 < nkv) load_tile(t + 1);    // overlap globals with compute

        if (kv0 > iMax) continue;             // tile fully masked for wave

        // ---- S = Q K^T : 8 WMMAs -----------------------------------------
        v8f s0 = {}, s1 = {}, s2 = {}, s3 = {};
#define BOP(jt, c) combine16(*(const v8h*)&Klds[(jt)*16 + ln][(c)*32 + half*8], \
                             *(const v8h*)&Klds[(jt)*16 + ln][(c)*32 + 16 + half*8])
        s0 = WMMA_F16(a0, BOP(0, 0), s0);  s0 = WMMA_F16(a1, BOP(0, 1), s0);
        s1 = WMMA_F16(a0, BOP(1, 0), s1);  s1 = WMMA_F16(a1, BOP(1, 1), s1);
        s2 = WMMA_F16(a0, BOP(2, 0), s2);  s2 = WMMA_F16(a1, BOP(2, 1), s2);
        s3 = WMMA_F16(a0, BOP(3, 0), s3);  s3 = WMMA_F16(a1, BOP(3, 1), s3);
#undef BOP

        // ---- online softmax ----------------------------------------------
        _Float16* Pw = &Plds[wave][0][0];     // row stride 72 halves
        const bool interior = (kv0 + KTILE) <= (iMin + 1);  // no masking needed

        auto srow = [&](int r, bool MASKED) {
            const int i = ibase + r;
            float t0 = s0[r], t1 = s1[r], t2 = s2[r], t3 = s3[r];
            if (MASKED) {   // replace == add here: -1e20 + O(1) == -1e20 in f32
                t0 = (kv0      + ln <= i) ? t0 : NEGV;
                t1 = (kv0 + 16 + ln <= i) ? t1 : NEGV;
                t2 = (kv0 + 32 + ln <= i) ? t2 : NEGV;
                t3 = (kv0 + 48 + ln <= i) ? t3 : NEGV;
            }
            t0 = (t0 == 0.0f) ? NEGV : t0;    // reference's exact-zero quirk
            t1 = (t1 == 0.0f) ? NEGV : t1;
            t2 = (t2 == 0.0f) ? NEGV : t2;
            t3 = (t3 == 0.0f) ? NEGV : t3;
            float tm = fmaxf(fmaxf(t0, t1), fmaxf(t2, t3));
#pragma unroll
            for (int d = 1; d < 16; d <<= 1) tm = fmaxf(tm, __shfl_xor(tm, d, 16));
            const float mn    = fmaxf(m[r], tm);
            const float alpha = __expf(m[r] - mn);
            m[r] = mn;
            o0[r] *= alpha; o1[r] *= alpha; o2[r] *= alpha; o3[r] *= alpha;
            lC[r] *= alpha;
            _Float16* prow = Pw + (half * 8 + r) * 72;
            prow[ln]      = (_Float16)__expf(t0 - mn);
            prow[16 + ln] = (_Float16)__expf(t1 - mn);
            prow[32 + ln] = (_Float16)__expf(t2 - mn);
            prow[48 + ln] = (_Float16)__expf(t3 - mn);
        };
        if (interior) {
#pragma unroll
            for (int r = 0; r < 8; ++r) srow(r, false);
        } else {
#pragma unroll
            for (int r = 0; r < 8; ++r) srow(r, true);
        }
        // wave-private P scratch, lanes in lockstep: LDS-counter order only
        asm volatile("s_wait_dscnt 0" ::: "memory");

        // ---- P as A-operand; l via ones-WMMA; O += P V : 10 WMMAs --------
        const _Float16* parow = Pw + ln * 72;
        v16h pa0 = combine16(*(const v8h*)(parow + half * 8),
                             *(const v8h*)(parow + 16 + half * 8));
        v16h pa1 = combine16(*(const v8h*)(parow + 32 + half * 8),
                             *(const v8h*)(parow + 48 + half * 8));
        lC = WMMA_F16(pa0, onesB, lC);        // lC[r] = sum_j P[row][j]
        lC = WMMA_F16(pa1, onesB, lC);
#define VOP(nt, c) combine16(*(const v8h*)&Vtlds[(nt)*16 + ln][(c)*32 + half*8], \
                             *(const v8h*)&Vtlds[(nt)*16 + ln][(c)*32 + 16 + half*8])
        o0 = WMMA_F16(pa0, VOP(0, 0), o0);  o0 = WMMA_F16(pa1, VOP(0, 1), o0);
        o1 = WMMA_F16(pa0, VOP(1, 0), o1);  o1 = WMMA_F16(pa1, VOP(1, 1), o1);
        o2 = WMMA_F16(pa0, VOP(2, 0), o2);  o2 = WMMA_F16(pa1, VOP(2, 1), o2);
        o3 = WMMA_F16(pa0, VOP(3, 0), o3);  o3 = WMMA_F16(pa1, VOP(3, 1), o3);
#undef VOP
    }

    // ---- epilogue: O / l -> global ---------------------------------------
    float* ob = out + ((size_t)b * S + (size_t)ibase) * D;
#pragma unroll
    for (int r = 0; r < 8; ++r) {
        const float inv = 1.0f / lC[r];
        float* orow = ob + r * D;
        orow[ln]      = o0[r] * inv;
        orow[16 + ln] = o1[r] * inv;
        orow[32 + ln] = o2[r] * inv;
        orow[48 + ln] = o3[r] * inv;
    }
}

extern "C" void kernel_launch(void* const* d_in, const int* in_sizes, int n_in,
                              void* d_out, int out_size, void* d_ws, size_t ws_size,
                              hipStream_t stream) {
    const float* q = (const float*)d_in[0];
    const float* k = (const float*)d_in[1];
    const float* v = (const float*)d_in[2];
    float* o = (float*)d_out;

    constexpr int B = 16, S = 2048, QTILE = 128;
    dim3 grid(B * (S / QTILE));   // 256 workgroups
    dim3 block(256);              // 8 wave32s
    fa_causal_kernel<<<grid, block, 0, stream>>>(q, k, v, o);
}
